// MinCutPoolModel_38981123178597
// MI455X (gfx1250) — matrix operand order repaired and checked by reading, not compiled, not run on hardware
//
#include <hip/hip_runtime.h>
#include <hip/hip_bf16.h>
#include <math.h>

typedef __attribute__((ext_vector_type(16))) _Float16 v16h;
typedef __attribute__((ext_vector_type(4)))  _Float16 v4h;
typedef __attribute__((ext_vector_type(8)))  float    v8f;

#define BB    64
#define NN    512
#define HIDC  64
#define K1C   256
#define K2C   128
#define OUTC  10

// =====================================================================
// Batched GEMM on the CDNA5 matrix core (v_wmma_f32_16x16x32_f16).
// C[b] (MxN) = op(A[b]) (MxK) * B[b] (KxN)  [+Cin] [+bias] [relu]
// Block = 128 threads = 4 waves covering a 64x64 C supertile:
//   - 32x64 B slab staged once per K-step (coalesced float4 loads),
//     shared by all 4 waves through LDS.
//   - each wave stages its own 16x32 A slab, then issues 4 back-to-back
//     WMMAs (one A fragment reused against 4 B fragments).
// Tiles are written to LDS already permuted into per-lane WMMA fragment
// order, so fragment reads are plain 32B ds_load_b128 pairs.
// Template params keep the inner loop branch-free.
// =====================================================================
template<bool TA, bool ACC, bool RELU, bool HASB>
__global__ __launch_bounds__(128)
void k_bmm(const float* __restrict__ A, const float* __restrict__ B,
           const float* __restrict__ bias, const float* __restrict__ Cin,
           float* __restrict__ C, int M, int N, int K,
           long sA, long sB, long sC, int lda, int ldb, int ldc)
{
    __shared__ _Float16 fragA[4][32][16];      // [wave][lane][elem]
    __shared__ _Float16 fragB[4][32][16];      // [ntile][lane][elem]
    const int tid  = threadIdx.x;
    const int w    = tid >> 5, lane = tid & 31;
    const int half = lane >> 4, lr = lane & 15;
    const float* Ab = A + (long)blockIdx.z * sA;
    const float* Bb = B + (long)blockIdx.z * sB;
    float*       Cb = C + (long)blockIdx.z * sC;
    const int m0 = (blockIdx.x * 4 + w) * 16;
    const int n0 = blockIdx.y * 64;

    v8f accv[4] = {{}, {}, {}, {}};
    for (int k0 = 0; k0 < K; k0 += 32) {
        if (k0 + 32 < K) {  // hint next K-slab toward the caches
            __builtin_prefetch(&Bb[(long)(k0 + 32) * ldb + n0], 0, 1);
            __builtin_prefetch(TA ? &Ab[(long)(k0 + 32) * lda + m0]
                                  : &Ab[(long)m0 * lda + (k0 + 32)], 0, 1);
        }
        // ---- stage B 32x64 slab, fragment-ordered: 128 threads x 4 float4
        #pragma unroll
        for (int j = 0; j < 4; ++j) {
            int q = tid + 128 * j;                           // 0..511 chunks
            int r = q >> 4, c0 = (q & 15) * 4;               // K-row, col base
            float4 v = *(const float4*)&Bb[(long)(k0 + r) * ldb + (n0 + c0)];
            int nt = c0 >> 4;                                // which 16-col tile
            int lb = (c0 & 15) + (((r >> 3) & 1) << 4);      // dest lane base
            int i  = (r & 7) | ((r >> 4) << 3);              // dest elem
            fragB[nt][lb + 0][i] = (_Float16)v.x;
            fragB[nt][lb + 1][i] = (_Float16)v.y;
            fragB[nt][lb + 2][i] = (_Float16)v.z;
            fragB[nt][lb + 3][i] = (_Float16)v.w;
        }
        // ---- stage this wave's A 16x32 slab, fragment-ordered
        if (!TA) {
            #pragma unroll
            for (int j = 0; j < 4; ++j) {
                int q = lane + 32 * j;                       // 0..127 chunks
                int r = q >> 3, c0 = (q & 7) * 4;            // row, col base
                float4 v = *(const float4*)&Ab[(long)(m0 + r) * lda + (k0 + c0)];
                int ld = r + (((c0 >> 3) & 1) << 4);         // dest lane
                int i0 = (c0 & 7) | ((c0 >> 4) << 3);        // dest elem (4 consecutive)
                v4h pk;
                pk.x = (_Float16)v.x; pk.y = (_Float16)v.y;
                pk.z = (_Float16)v.z; pk.w = (_Float16)v.w;
                *(v4h*)&fragA[w][ld][i0] = pk;               // one ds_store_b64
            }
        } else {
            #pragma unroll
            for (int j = 0; j < 4; ++j) {
                int q = lane + 32 * j;
                int c = q >> 2, r0 = (q & 3) * 4;            // k-col, 4 rows (contig in mem)
                float4 v = *(const float4*)&Ab[(long)(k0 + c) * lda + (m0 + r0)];
                int i  = (c & 7) | ((c >> 4) << 3);
                int hb = ((c >> 3) & 1) << 4;
                fragA[w][r0 + 0 + hb][i] = (_Float16)v.x;
                fragA[w][r0 + 1 + hb][i] = (_Float16)v.y;
                fragA[w][r0 + 2 + hb][i] = (_Float16)v.z;
                fragA[w][r0 + 3 + hb][i] = (_Float16)v.w;
            }
        }
        __syncthreads();
        v16h a = *(const v16h*)&fragA[w][lane][0];           // 2x ds_load_b128
        #pragma unroll
        for (int nt = 0; nt < 4; ++nt) {                     // 4 back-to-back WMMAs
            v16h b = *(const v16h*)&fragB[nt][lane][0];
            accv[nt] = __builtin_amdgcn_wmma_f32_16x16x32_f16(
                           false, a, false, b, (short)0, accv[nt], false, false);
        }
        __syncthreads();
    }

    #pragma unroll
    for (int nt = 0; nt < 4; ++nt) {
        union { v8f v; float e[8]; } uc; uc.v = accv[nt];
        #pragma unroll
        for (int j = 0; j < 8; ++j) {
            int row = m0 + j + half * 8;
            int col = n0 + nt * 16 + lr;
            float v = uc.e[j];
            if constexpr (ACC)  v += Cin[(long)blockIdx.z * sC + (long)row * ldc + col];
            if constexpr (HASB) v += bias[col];
            if constexpr (RELU) v = fmaxf(v, 0.f);
            Cb[(long)row * ldc + col] = v;
        }
    }
}

// =====================================================================
// Utility kernels (latency/atomic-bound parts of the graph pipeline)
// =====================================================================
__global__ void k_fill(float* p, float v, long n) {
    long i = (long)blockIdx.x * blockDim.x + threadIdx.x;
    if (i < n) p[i] = v;
}

__global__ void k_deg(const int* __restrict__ dst, float* deg, int E) {
    int e = blockIdx.x * blockDim.x + threadIdx.x;
    if (e < E) atomicAdd(&deg[dst[e]], 1.0f);
}

__global__ void k_dinv(const float* deg, float* dinv, int n) {
    int i = blockIdx.x * blockDim.x + threadIdx.x;
    if (i < n) dinv[i] = rsqrtf(deg[i]);
}

__global__ void k_gcn_scatter(const int* __restrict__ src, const int* __restrict__ dst,
                              const float* __restrict__ dinv, const float* __restrict__ xw,
                              float* acc, long total) {
    long idx = (long)blockIdx.x * blockDim.x + threadIdx.x;
    if (idx >= total) return;
    int e = (int)(idx >> 6), f = (int)(idx & 63);
    int s = src[e], d = dst[e];
    atomicAdd(&acc[(long)d * HIDC + f], dinv[s] * dinv[d] * xw[(long)s * HIDC + f]);
}

__global__ void k_gcn_final(float* h, const float* xw, const float* dinv,
                            const float* b1, long total) {
    long idx = (long)blockIdx.x * blockDim.x + threadIdx.x;
    if (idx >= total) return;
    long i = idx >> 6; int f = (int)(idx & 63);
    float di = dinv[i];
    h[idx] = fmaxf(h[idx] + di * di * xw[idx] + b1[f], 0.f);
}

__global__ void k_adj_build(const int* __restrict__ src, const int* __restrict__ dst,
                            float* adj, int E) {
    int e = blockIdx.x * blockDim.x + threadIdx.x;
    if (e >= E) return;
    int s = src[e], d = dst[e];
    int g = s / NN;
    atomicAdd(&adj[(long)g * NN * NN + (long)(s - g * NN) * NN + (d - g * NN)], 1.0f);
}

__global__ void k_softmax(float* s, int cols) {   // one 128-thread block per row
    __shared__ float red[128];
    float* p = s + (long)blockIdx.x * cols;
    int t = threadIdx.x;
    float mx = -1e30f;
    for (int c = t; c < cols; c += 128) mx = fmaxf(mx, p[c]);
    red[t] = mx; __syncthreads();
    for (int o = 64; o > 0; o >>= 1) { if (t < o) red[t] = fmaxf(red[t], red[t + o]); __syncthreads(); }
    mx = red[0]; __syncthreads();
    float sum = 0.f;
    for (int c = t; c < cols; c += 128) { float e = expf(p[c] - mx); p[c] = e; sum += e; }
    red[t] = sum; __syncthreads();
    for (int o = 64; o > 0; o >>= 1) { if (t < o) red[t] += red[t + o]; __syncthreads(); }
    float inv = 1.f / red[0];
    for (int c = t; c < cols; c += 128) p[c] *= inv;
}

__global__ void k_rowsum(const float* a, float* out, int m, long rows) {
    long r = (long)blockIdx.x * blockDim.x + threadIdx.x;
    if (r >= rows) return;
    const float* p = a + r * (long)m;
    float s = 0.f;
    for (int j = 0; j < m; ++j) s += p[j];
    out[r] = s;
}

__global__ void k_pool_loss(const float* __restrict__ out_adj, const float* __restrict__ ss,
                            const float* __restrict__ s, const float* __restrict__ d_flat,
                            float* loss, int n, int k)
{
    __shared__ float red[256];
    int b = blockIdx.x, t = threadIdx.x;
    const float* oa  = out_adj + (long)b * k * k;
    const float* sb  = s       + (long)b * n * k;
    const float* ssb = ss      + (long)b * k * k;
    const float* db  = d_flat  + (long)b * n;

    float num = 0.f;
    for (int i = t; i < k; i += 256) num += oa[(long)i * k + i];
    red[t] = num; __syncthreads();
    for (int o = 128; o > 0; o >>= 1) { if (t < o) red[t] += red[t + o]; __syncthreads(); }
    num = red[0]; __syncthreads();

    float den = 0.f;
    for (long i = t; i < (long)n * k; i += 256) { float sv = sb[i]; den += db[i / k] * sv * sv; }
    red[t] = den; __syncthreads();
    for (int o = 128; o > 0; o >>= 1) { if (t < o) red[t] += red[t + o]; __syncthreads(); }
    den = red[0]; __syncthreads();

    float f2 = 0.f;
    for (long i = t; i < (long)k * k; i += 256) { float v = ssb[i]; f2 += v * v; }
    red[t] = f2; __syncthreads();
    for (int o = 128; o > 0; o >>= 1) { if (t < o) red[t] += red[t + o]; __syncthreads(); }
    float fro = sqrtf(red[0]); __syncthreads();

    float inv_sk = rsqrtf((float)k);
    float dsum = 0.f;
    for (long i = t; i < (long)k * k; i += 256) {
        float diag = ((i / k) == (i % k)) ? inv_sk : 0.f;
        float d = ssb[i] / fro - diag;
        dsum += d * d;
    }
    red[t] = dsum; __syncthreads();
    for (int o = 128; o > 0; o >>= 1) { if (t < o) red[t] += red[t + o]; __syncthreads(); }
    if (t == 0) atomicAdd(loss, (-(num / den) + sqrtf(red[0])) * (1.0f / BB));
}

__global__ void k_zero_diag(float* a, int k) {
    int i = blockIdx.x * blockDim.x + threadIdx.x;
    if (i >= BB * k) return;
    int b = i / k, d = i % k;
    a[(long)b * k * k + (long)d * k + d] = 0.f;
}

__global__ void k_adj_dvec(const float* a, float* dv, int k) {
    int i = blockIdx.x * blockDim.x + threadIdx.x;
    if (i >= BB * k) return;
    const float* p = a + (long)i * k;
    float s = 0.f;
    for (int j = 0; j < k; ++j) s += p[j];
    dv[i] = sqrtf(s) + 1e-15f;
}

__global__ void k_adj_norm(float* a, const float* dv, int k) {
    long i = (long)blockIdx.x * blockDim.x + threadIdx.x;
    if (i >= (long)BB * k * k) return;
    long b = i / ((long)k * k);
    long rc = i % ((long)k * k);
    int r = (int)(rc / k), c = (int)(rc % k);
    a[i] /= dv[b * k + r] * dv[b * k + c];
}

__global__ void k_mean(const float* x, float* xg, int n) {
    int i = blockIdx.x * blockDim.x + threadIdx.x;
    if (i >= BB * HIDC) return;
    int b = i / HIDC, f = i % HIDC;
    const float* p = x + (long)b * n * HIDC + f;
    float s = 0.f;
    for (int j = 0; j < n; ++j) s += p[(long)j * HIDC];
    xg[i] = s * (1.0f / n);
}

__global__ void k_head(const float* __restrict__ hg, const float* __restrict__ Wl2,
                       const float* __restrict__ bl2, float* out)
{
    __shared__ float lg[OUTC];
    __shared__ float mls[2];
    int b = blockIdx.x, t = threadIdx.x;
    if (t < OUTC) {
        float s = bl2[t];
        for (int j = 0; j < HIDC; ++j) s += hg[b * HIDC + j] * Wl2[j * OUTC + t];
        lg[t] = s;
    }
    __syncthreads();
    if (t == 0) {
        float mx = lg[0];
        for (int j = 1; j < OUTC; ++j) mx = fmaxf(mx, lg[j]);
        float se = 0.f;
        for (int j = 0; j < OUTC; ++j) se += expf(lg[j] - mx);
        mls[0] = mx; mls[1] = logf(se);
    }
    __syncthreads();
    if (t < OUTC) out[b * OUTC + t] = lg[t] - mls[0] - mls[1];
}

__global__ void k_copy1(const float* src, float* dst) { dst[0] = src[0]; }

// =====================================================================
// Host orchestration
// =====================================================================
enum BmmMode { BM_PLAIN, BM_BIAS, BM_ACC_RELU, BM_ACC, BM_BIAS_RELU, BM_TA };

static inline void bmm(hipStream_t st, BmmMode mode,
                       const float* A, const float* B, const float* bias,
                       const float* Cin, float* C, int M, int N, int K,
                       long sA, long sB, long sC, int lda, int ldb, int ldc, int batch)
{
    dim3 g(M / 64, N / 64, batch);
    switch (mode) {
    case BM_PLAIN:
        k_bmm<false,false,false,false><<<g,128,0,st>>>(A,B,bias,Cin,C,M,N,K,sA,sB,sC,lda,ldb,ldc); break;
    case BM_BIAS:
        k_bmm<false,false,false,true ><<<g,128,0,st>>>(A,B,bias,Cin,C,M,N,K,sA,sB,sC,lda,ldb,ldc); break;
    case BM_ACC_RELU:
        k_bmm<false,true, true, false><<<g,128,0,st>>>(A,B,bias,Cin,C,M,N,K,sA,sB,sC,lda,ldb,ldc); break;
    case BM_ACC:
        k_bmm<false,true, false,false><<<g,128,0,st>>>(A,B,bias,Cin,C,M,N,K,sA,sB,sC,lda,ldb,ldc); break;
    case BM_BIAS_RELU:
        k_bmm<false,false,true, true ><<<g,128,0,st>>>(A,B,bias,Cin,C,M,N,K,sA,sB,sC,lda,ldb,ldc); break;
    case BM_TA:
        k_bmm<true, false,false,false><<<g,128,0,st>>>(A,B,bias,Cin,C,M,N,K,sA,sB,sC,lda,ldb,ldc); break;
    }
}
static inline int div_up(long a, int b) { return (int)((a + b - 1) / b); }

extern "C" void kernel_launch(void* const* d_in, const int* in_sizes, int n_in,
                              void* d_out, int out_size, void* d_ws, size_t ws_size,
                              hipStream_t stream)
{
    const float* x     = (const float*)d_in[0];
    const int*   ei    = (const int*)  d_in[1];
    const float* W1    = (const float*)d_in[3];
    const float* b1    = (const float*)d_in[4];
    const float* Wm1   = (const float*)d_in[5];
    const float* bm1   = (const float*)d_in[6];
    const float* Wrel2 = (const float*)d_in[7];
    const float* Wroot2= (const float*)d_in[8];
    const float* b2    = (const float*)d_in[9];
    const float* Wm2   = (const float*)d_in[10];
    const float* bm2   = (const float*)d_in[11];
    const float* Wrel3 = (const float*)d_in[12];
    const float* Wroot3= (const float*)d_in[13];
    const float* b3    = (const float*)d_in[14];
    const float* Wl1   = (const float*)d_in[15];
    const float* bl1   = (const float*)d_in[16];
    const float* Wl2   = (const float*)d_in[17];
    const float* bl2   = (const float*)d_in[18];

    const int E    = in_sizes[1] / 2;
    const int Ntot = in_sizes[0] / 128;          // 32768
    const int* src = ei;
    const int* dst = ei + E;

    // ---- workspace layout (bump allocation with explicit reuse; peak ~145MB)
    char* ws = (char*)d_ws;
    const long MBy = 1 << 20;
    float* xw    = (float*)(ws + 0);                       // 8 MB [32768,64]
    float* h     = (float*)(ws + 8  * MBy);                // 8 MB [64,512,64]
    float* deg   = (float*)(ws + 16 * MBy);                // 256 KB slot (reused: dflat1/dflat2)
    float* dinv  = (float*)(ws + 16 * MBy + 256 * 1024);   // 256 KB slot (reused: dvec1/dvec2)
    float* loss  = (float*)(ws + 16 * MBy + 512 * 1024);   // scalar
    float* adj   = (float*)(ws + 17 * MBy);                // 64 MB [64,512,512]
    float* s1    = (float*)(ws + 81 * MBy);                // 32 MB [64,512,256]
    float* as1   = (float*)(ws + 113 * MBy);               // 32 MB [64,512,256]
    // adj region reuse (after A@s and degree rowsums):
    float* oadj1 = adj;                                    // 16 MB [64,256,256]
    float* ss1   = (float*)(ws + 33 * MBy);                // 16 MB
    float* xd2   = (float*)(ws + 49 * MBy);                // 4 MB  [64,256,64]
    // s1 region reuse (stage 2):
    float* t2    = s1;                                     // 4 MB
    float* t2b   = (float*)(ws + 85 * MBy);                // 4 MB
    float* xd2b  = (float*)(ws + 89 * MBy);                // 4 MB
    float* s2    = (float*)(ws + 93 * MBy);                // 8 MB [64,256,128]
    float* as2   = (float*)(ws + 101 * MBy);               // 8 MB
    float* oadj2 = (float*)(ws + 109 * MBy);               // 4 MB [64,128,128]
    // as1 region reuse (stage 2/3):
    float* ss2   = as1;                                    // 4 MB
    float* xd3   = (float*)(ws + 117 * MBy);               // 2 MB [64,128,64]
    float* t3    = (float*)(ws + 119 * MBy);               // 2 MB
    float* xd3b  = (float*)(ws + 121 * MBy);               // 2 MB
    float* xg    = (float*)(ws + 123 * MBy);               // 16 KB [64,64]
    float* hg    = (float*)(ws + 123 * MBy + 64 * 1024);   // 16 KB
    float* dflat = deg;   // [64,512] then [64,256]
    float* dvec  = dinv;  // [64,256] then [64,128]

    float* outp = (float*)d_out;

    // ---- GCNConv -----------------------------------------------------
    bmm(stream, BM_PLAIN, x, W1, nullptr, nullptr, xw, Ntot, HIDC, 128,
        0, 0, 0, 128, HIDC, HIDC, 1);
    k_fill<<<div_up(Ntot, 256), 256, 0, stream>>>(deg, 1.0f, Ntot);  // self loop
    k_deg<<<div_up(E, 256), 256, 0, stream>>>(dst, deg, E);
    k_dinv<<<div_up(Ntot, 256), 256, 0, stream>>>(deg, dinv, Ntot);
    k_fill<<<div_up((long)Ntot * HIDC, 256), 256, 0, stream>>>(h, 0.0f, (long)Ntot * HIDC);
    k_gcn_scatter<<<div_up((long)E * HIDC, 256), 256, 0, stream>>>(src, dst, dinv, xw, h, (long)E * HIDC);
    k_gcn_final<<<div_up((long)Ntot * HIDC, 256), 256, 0, stream>>>(h, xw, dinv, b1, (long)Ntot * HIDC);

    // ---- dense adjacency + loss accumulator --------------------------
    k_fill<<<div_up((long)BB * NN * NN, 256), 256, 0, stream>>>(adj, 0.0f, (long)BB * NN * NN);
    k_adj_build<<<div_up(E, 256), 256, 0, stream>>>(src, dst, adj, E);
    k_fill<<<1, 32, 0, stream>>>(loss, 0.0f, 8);

    // ---- pool 1 ------------------------------------------------------
    bmm(stream, BM_BIAS, h, Wm1, bm1, nullptr, s1, NN, K1C, HIDC,
        (long)NN * HIDC, 0, (long)NN * K1C, HIDC, K1C, K1C, BB);
    k_softmax<<<BB * NN, 128, 0, stream>>>(s1, K1C);
    bmm(stream, BM_PLAIN, adj, s1, nullptr, nullptr, as1, NN, K1C, NN,
        (long)NN * NN, (long)NN * K1C, (long)NN * K1C, NN, K1C, K1C, BB);
    k_rowsum<<<div_up((long)BB * NN, 256), 256, 0, stream>>>(adj, dflat, NN, (long)BB * NN);
    // adj region now dead -> reuse for out_adj1 / ss1 / xd2
    bmm(stream, BM_TA, s1, as1, nullptr, nullptr, oadj1, K1C, K1C, NN,
        (long)NN * K1C, (long)NN * K1C, (long)K1C * K1C, K1C, K1C, K1C, BB);
    bmm(stream, BM_TA, s1, s1, nullptr, nullptr, ss1, K1C, K1C, NN,
        (long)NN * K1C, (long)NN * K1C, (long)K1C * K1C, K1C, K1C, K1C, BB);
    bmm(stream, BM_TA, s1, h, nullptr, nullptr, xd2, K1C, HIDC, NN,
        (long)NN * K1C, (long)NN * HIDC, (long)K1C * HIDC, K1C, HIDC, HIDC, BB);
    k_pool_loss<<<BB, 256, 0, stream>>>(oadj1, ss1, s1, dflat, loss, NN, K1C);
    k_zero_diag<<<div_up(BB * K1C, 256), 256, 0, stream>>>(oadj1, K1C);
    k_adj_dvec<<<div_up(BB * K1C, 256), 256, 0, stream>>>(oadj1, dvec, K1C);
    k_adj_norm<<<div_up((long)BB * K1C * K1C, 256), 256, 0, stream>>>(oadj1, dvec, K1C);

    // ---- DenseGraphConv 2: relu(A x Wrel + x Wroot + b) --------------
    bmm(stream, BM_PLAIN, oadj1, xd2, nullptr, nullptr, t2, K1C, HIDC, K1C,
        (long)K1C * K1C, (long)K1C * HIDC, (long)K1C * HIDC, K1C, HIDC, HIDC, BB);
    bmm(stream, BM_BIAS, xd2, Wroot2, b2, nullptr, t2b, K1C, HIDC, HIDC,
        (long)K1C * HIDC, 0, (long)K1C * HIDC, HIDC, HIDC, HIDC, BB);
    bmm(stream, BM_ACC_RELU, t2, Wrel2, nullptr, t2b, xd2b, K1C, HIDC, HIDC,
        (long)K1C * HIDC, 0, (long)K1C * HIDC, HIDC, HIDC, HIDC, BB);

    // ---- pool 2 ------------------------------------------------------
    bmm(stream, BM_BIAS, xd2b, Wm2, bm2, nullptr, s2, K1C, K2C, HIDC,
        (long)K1C * HIDC, 0, (long)K1C * K2C, HIDC, K2C, K2C, BB);
    k_softmax<<<BB * K1C, 128, 0, stream>>>(s2, K2C);
    bmm(stream, BM_PLAIN, oadj1, s2, nullptr, nullptr, as2, K1C, K2C, K1C,
        (long)K1C * K1C, (long)K1C * K2C, (long)K1C * K2C, K1C, K2C, K2C, BB);
    k_rowsum<<<div_up((long)BB * K1C, 256), 256, 0, stream>>>(oadj1, dflat, K1C, (long)BB * K1C);
    bmm(stream, BM_TA, s2, as2, nullptr, nullptr, oadj2, K2C, K2C, K1C,
        (long)K1C * K2C, (long)K1C * K2C, (long)K2C * K2C, K2C, K2C, K2C, BB);
    bmm(stream, BM_TA, s2, s2, nullptr, nullptr, ss2, K2C, K2C, K1C,
        (long)K1C * K2C, (long)K1C * K2C, (long)K2C * K2C, K2C, K2C, K2C, BB);
    bmm(stream, BM_TA, s2, xd2b, nullptr, nullptr, xd3, K2C, HIDC, K1C,
        (long)K1C * K2C, (long)K1C * HIDC, (long)K2C * HIDC, K2C, HIDC, HIDC, BB);
    k_pool_loss<<<BB, 256, 0, stream>>>(oadj2, ss2, s2, dflat, loss, K1C, K2C);
    k_zero_diag<<<div_up(BB * K2C, 256), 256, 0, stream>>>(oadj2, K2C);
    k_adj_dvec<<<div_up(BB * K2C, 256), 256, 0, stream>>>(oadj2, dvec, K2C);
    k_adj_norm<<<div_up((long)BB * K2C * K2C, 256), 256, 0, stream>>>(oadj2, dvec, K2C);

    // ---- DenseGraphConv 3 (no relu) ----------------------------------
    bmm(stream, BM_PLAIN, oadj2, xd3, nullptr, nullptr, t3, K2C, HIDC, K2C,
        (long)K2C * K2C, (long)K2C * HIDC, (long)K2C * HIDC, K2C, HIDC, HIDC, BB);
    bmm(stream, BM_BIAS, xd3, Wroot3, b3, nullptr, xd3b, K2C, HIDC, HIDC,
        (long)K2C * HIDC, 0, (long)K2C * HIDC, HIDC, HIDC, HIDC, BB);
    bmm(stream, BM_ACC, t3, Wrel3, nullptr, xd3b, xd3b, K2C, HIDC, HIDC,
        (long)K2C * HIDC, 0, (long)K2C * HIDC, HIDC, HIDC, HIDC, BB);

    // ---- readout + head ---------------------------------------------
    k_mean<<<div_up(BB * HIDC, 256), 256, 0, stream>>>(xd3b, xg, K2C);
    bmm(stream, BM_BIAS_RELU, xg, Wl1, bl1, nullptr, hg, BB, HIDC, HIDC,
        0, 0, 0, HIDC, HIDC, HIDC, 1);
    k_head<<<BB, 32, 0, stream>>>(hg, Wl2, bl2, outp);
    k_copy1<<<1, 1, 0, stream>>>(loss, outp + BB * OUTC);
}
